// adapter_43911745634779
// MI455X (gfx1250) — matrix operand order repaired and checked by reading, not compiled, not run on hardware
//
#include <hip/hip_runtime.h>
#include <hip/hip_bf16.h>

typedef __attribute__((ext_vector_type(16))) _Float16 v16h;
typedef __attribute__((ext_vector_type(4)))  _Float16 v4h;
typedef __attribute__((ext_vector_type(8)))  float    v8f;

#define C_DIM 768
#define H_DIM 64
#define MT 64            // tokens per tile
#define KT1 (C_DIM/32)   // 24 k-tiles for GEMM1
#define NT1 (H_DIM/16)   // 4  n-tiles for GEMM1
#define NT2 (C_DIM/16)   // 48 n-tiles for GEMM2

// LDS layout (bytes)
#define W1_OFF   0
#define W2_OFF   (W1_OFF + C_DIM*H_DIM*2)       // 96 KB f16 swizzled w1
#define XN_OFF   (W2_OFF + H_DIM*C_DIM*2)       // 96 KB f16 swizzled w2
#define HH_OFF   (XN_OFF + MT*C_DIM*2)          // 96 KB f16 normalized x tile
#define B1_OFF   (HH_OFF + MT*H_DIM*2)          // 8 KB f16 hidden tile
#define B2_OFF   (B1_OFF + H_DIM*4)
#define GM_OFF   (B2_OFF + C_DIM*4)
#define BT_OFF   (GM_OFF + C_DIM*4)
#define LDS_BYTES (BT_OFF + C_DIM*4)            // = 312576 B < 320 KB WGP LDS

union FragH { v16h v; uint4 q[2]; };

// A fragment (16x32 f16) from row-major LDS buffer.
// Lane l: M = l&15, K-group = l>>4. VGPR0-3 = K[kg*8 .. kg*8+7] (contiguous 16B),
// VGPR4-7 = K[16+kg*8 .. 16+kg*8+7] (contiguous 16B).  -> two ds_load_b128.
__device__ inline v16h load_a_frag(const _Float16* buf, int rowStride, int row0, int k0) {
    int lane = threadIdx.x & 31;
    const _Float16* p = buf + (row0 + (lane & 15)) * rowStride + k0 + (lane >> 4) * 8;
    FragH f;
    f.q[0] = *(const uint4*)(p);
    f.q[1] = *(const uint4*)(p + 16);
    return f.v;
}

// B fragment (32x16 f16) from the pre-swizzled weight buffer: 16 halves per lane
// stored contiguously per (ktile, ntile, lane). -> two ds_load_b128.
__device__ inline v16h load_b_frag(const _Float16* wsw, int NT, int kt, int nt) {
    int lane = threadIdx.x & 31;
    const _Float16* p = wsw + ((((kt * NT + nt) << 5) + lane) << 4);
    FragH f;
    f.q[0] = *(const uint4*)(p);
    f.q[1] = *(const uint4*)(p + 8);
    return f.v;
}

// Stage an f32 [K][Nw] row-major weight matrix into LDS as f16 B-fragments:
// lane l of fragment (kt,nt) holds column n = nt*16+(l&15), K = kt*32+(l>>4)*8+e (e<8), kt*32+16+(l>>4)*8+(e-8) (e>=8).
__device__ inline void stage_weights(const float* __restrict__ w, int K, int Nw, int NT,
                                     _Float16* dst) {
    int nchunks = (K / 32) * NT * 32;
    for (int idx = threadIdx.x; idx < nchunks; idx += blockDim.x) {
        int lane = idx & 31;
        int t    = idx >> 5;
        int nt   = t % NT;
        int kt   = t / NT;
        int n     = nt * 16 + (lane & 15);
        int kbase = kt * 32 + (lane >> 4) * 8;
        _Float16* d = dst + idx * 16;
#pragma unroll
        for (int e = 0; e < 16; ++e)
            d[e] = (_Float16)w[(kbase + e + ((e >> 3) << 3)) * Nw + n];
    }
}

__global__ void __launch_bounds__(256)
adapter_ln_mlp_kernel(const float* __restrict__ x,  const float* __restrict__ w1,
                      const float* __restrict__ b1, const float* __restrict__ w2,
                      const float* __restrict__ b2, const float* __restrict__ gamma,
                      const float* __restrict__ beta, float* __restrict__ out, int ntok) {
    extern __shared__ char smem[];
    _Float16* w1s = (_Float16*)(smem + W1_OFF);
    _Float16* w2s = (_Float16*)(smem + W2_OFF);
    _Float16* xnh = (_Float16*)(smem + XN_OFF);
    _Float16* hh  = (_Float16*)(smem + HH_OFF);
    float* b1f = (float*)(smem + B1_OFF);
    float* b2f = (float*)(smem + B2_OFF);
    float* gm  = (float*)(smem + GM_OFF);
    float* bt  = (float*)(smem + BT_OFF);

    const int tid  = threadIdx.x;
    const int wave = tid >> 5;
    const int lane = tid & 31;

    // ---- Stage weights (once per persistent WG) + small vectors ----
    stage_weights(w1, C_DIM, H_DIM, NT1, w1s);
    stage_weights(w2, H_DIM, C_DIM, NT2, w2s);
    for (int i = tid; i < H_DIM; i += blockDim.x) b1f[i] = b1[i];
    for (int i = tid; i < C_DIM; i += blockDim.x) {
        b2f[i] = b2[i]; gm[i] = gamma[i]; bt[i] = beta[i];
    }
    __syncthreads();

    const int ntiles = ntok / MT;
    for (int tile = blockIdx.x; tile < ntiles; tile += gridDim.x) {
        const int base = tile * MT;

        // prefetch next tile of x into the cache hierarchy (global_prefetch_b8)
        int nxt = tile + gridDim.x;
        if (nxt < ntiles)
            __builtin_prefetch(x + (size_t)nxt * MT * C_DIM + tid * 192, 0, 0);

        // ---- LayerNorm: 8 tokens per wave, f32, wave32 reduction ----
        for (int i = 0; i < 8; ++i) {
            int t = wave * 8 + i;
            const float* xg = x + (size_t)(base + t) * C_DIM;
            float4 vbuf[6];
            float s = 0.f, s2 = 0.f;
#pragma unroll
            for (int c = 0; c < 6; ++c) {
                float4 v = *(const float4*)(xg + c * 128 + lane * 4);
                vbuf[c] = v;
                s  += v.x + v.y + v.z + v.w;
                s2 += v.x * v.x + v.y * v.y + v.z * v.z + v.w * v.w;
            }
#pragma unroll
            for (int o = 16; o > 0; o >>= 1) {
                s  += __shfl_xor(s,  o, 32);
                s2 += __shfl_xor(s2, o, 32);
            }
            float mu   = s * (1.f / C_DIM);
            float var  = s2 * (1.f / C_DIM) - mu * mu;
            float rstd = rsqrtf(var + 1e-5f);
#pragma unroll
            for (int c = 0; c < 6; ++c) {
                int k = c * 128 + lane * 4;
                float4 v = vbuf[c];
                v4h hv;
                hv[0] = (_Float16)((v.x - mu) * rstd * gm[k + 0] + bt[k + 0]);
                hv[1] = (_Float16)((v.y - mu) * rstd * gm[k + 1] + bt[k + 1]);
                hv[2] = (_Float16)((v.z - mu) * rstd * gm[k + 2] + bt[k + 2]);
                hv[3] = (_Float16)((v.w - mu) * rstd * gm[k + 3] + bt[k + 3]);
                *(v4h*)(xnh + t * C_DIM + k) = hv;
            }
        }
        __syncthreads();

        // ---- GEMM1 [64x768]x[768x64] + bias + quickGELU -> hh (f16) ----
        {
            const int mt  = wave >> 1;
            const int nt0 = (wave & 1) * 2;
            v8f acc0 = {}, acc1 = {};
            for (int kt = 0; kt < KT1; ++kt) {
                v16h a  = load_a_frag(xnh, C_DIM, mt * 16, kt * 32);
                v16h bf0 = load_b_frag(w1s, NT1, kt, nt0);
                v16h bf1 = load_b_frag(w1s, NT1, kt, nt0 + 1);
                acc0 = __builtin_amdgcn_wmma_f32_16x16x32_f16(false, a, false, bf0,
                                                              (short)0, acc0, false, false);
                acc1 = __builtin_amdgcn_wmma_f32_16x16x32_f16(false, a, false, bf1,
                                                              (short)0, acc1, false, false);
            }
            // C layout: VGPR r, lanes 0-15 -> M=r, lanes 16-31 -> M=r+8; N = lane&15
            int mrow = mt * 16 + ((lane >> 4) << 3);
            int n0   = nt0 * 16 + (lane & 15);
#pragma unroll
            for (int r = 0; r < 8; ++r) {
                float h0 = acc0[r] + b1f[n0];
                float h1 = acc1[r] + b1f[n0 + 16];
                h0 = h0 * (1.f / (1.f + __expf(-1.702f * h0)));
                h1 = h1 * (1.f / (1.f + __expf(-1.702f * h1)));
                hh[(mrow + r) * H_DIM + n0]      = (_Float16)h0;
                hh[(mrow + r) * H_DIM + n0 + 16] = (_Float16)h1;
            }
        }
        __syncthreads();

        // ---- GEMM2 [64x64]x[64x768] + bias + residual -> out ----
        {
            const int mt = wave >> 1;
            v16h a0 = load_a_frag(hh, H_DIM, mt * 16, 0);
            v16h a1 = load_a_frag(hh, H_DIM, mt * 16, 32);
            const int ntbase = (wave & 1) * 24;
            for (int j = 0; j < 24; ++j) {
                int nt = ntbase + j;
                v16h bb0 = load_b_frag(w2s, NT2, 0, nt);
                v16h bb1 = load_b_frag(w2s, NT2, 1, nt);
                v8f acc = {};
                acc = __builtin_amdgcn_wmma_f32_16x16x32_f16(false, a0, false, bb0,
                                                             (short)0, acc, false, false);
                acc = __builtin_amdgcn_wmma_f32_16x16x32_f16(false, a1, false, bb1,
                                                             (short)0, acc, false, false);
                int col  = nt * 16 + (lane & 15);
                int mrow = base + mt * 16 + ((lane >> 4) << 3);
                float bias = b2f[col];
#pragma unroll
                for (int r = 0; r < 8; ++r) {
                    size_t idx = (size_t)(mrow + r) * C_DIM + col;
                    out[idx] = x[idx] + acc[r] + bias;
                }
            }
        }
        __syncthreads();  // protect xnh/hh reuse next iteration
    }
}

extern "C" void kernel_launch(void* const* d_in, const int* in_sizes, int n_in,
                              void* d_out, int out_size, void* d_ws, size_t ws_size,
                              hipStream_t stream) {
    (void)n_in; (void)out_size; (void)d_ws; (void)ws_size;
    const float* x     = (const float*)d_in[0];
    const float* w1    = (const float*)d_in[1];
    const float* b1    = (const float*)d_in[2];
    const float* w2    = (const float*)d_in[3];
    const float* b2    = (const float*)d_in[4];
    const float* gamma = (const float*)d_in[5];
    const float* beta  = (const float*)d_in[6];
    float* out = (float*)d_out;

    int ntok   = in_sizes[0] / C_DIM;   // 16*64*64 = 65536
    int ntiles = ntok / MT;

    (void)hipFuncSetAttribute((const void*)adapter_ln_mlp_kernel,
                              hipFuncAttributeMaxDynamicSharedMemorySize, LDS_BYTES);

    int grid = ntiles < 512 ? ntiles : 512;
    adapter_ln_mlp_kernel<<<grid, 256, LDS_BYTES, stream>>>(x, w1, b1, w2, b2, gamma,
                                                            beta, out, ntok);
}
